// ConformerMHSARelPosV1_292057776874
// MI455X (gfx1250) — matrix operand order repaired and checked by reading, not compiled, not run on hardware
//
#include <hip/hip_runtime.h>
#include <hip/hip_bf16.h>
#include <math.h>

typedef _Float16 f16;
typedef unsigned int u32;
typedef __attribute__((ext_vector_type(16))) _Float16 v16h;
typedef __attribute__((ext_vector_type(8)))  _Float16 v8h;
typedef __attribute__((ext_vector_type(8)))  float    v8f;
typedef __attribute__((ext_vector_type(4)))  unsigned int u32x4;
typedef __attribute__((ext_vector_type(8)))  int i32x8;
typedef __attribute__((ext_vector_type(4)))  int i32x4;

#define B_    8
#define T_    1024
#define F_    512
#define H_    8
#define D_    64
#define NREL  65          // 2*CLIP+1
#define BT    (B_*T_)     // 8192
#define BH    (B_*H_)     // 64

// Build a 16-half fragment from two contiguous 8-half (16B) chunks.
static __device__ __forceinline__ v16h load_frag(const f16* __restrict__ p0,
                                                 const f16* __restrict__ p1) {
  v8h lo = *(const v8h*)p0;
  v8h hi = *(const v8h*)p1;
  v16h r;
#pragma unroll
  for (int i = 0; i < 8; ++i) { r[i] = lo[i]; r[8 + i] = hi[i]; }
  return r;
}

static __device__ __forceinline__ v8f wmma_f16(v16h a, v16h b, v8f c) {
  return __builtin_amdgcn_wmma_f32_16x16x32_f16(false, a, false, b,
                                                (short)0, c, false, false);
}

// TDM: stage a 2D tile (f16 elements) from global into LDS.
// D# layout per CDNA5 ISA (2D: groups 2/3 zero).
static __device__ __forceinline__ void tdm_load_2d(u32 lds_off, const void* gptr,
                                                   u32 tensor_d0, u32 tensor_d1,
                                                   u32 stride0, u32 tile_d0, u32 tile_d1) {
  unsigned long long ga = (unsigned long long)gptr;
  u32x4 g0;
  g0.x = 1u;                                              // count=1, user descriptor
  g0.y = lds_off;                                         // lds_addr (bytes)
  g0.z = (u32)ga;                                         // global_addr[31:0]
  g0.w = (u32)((ga >> 32) & 0x1FFFFFFull) | (2u << 30);   // global_addr[56:32] | type=2
  i32x8 g1;
  g1[0] = (int)(1u << 16);                                // data_size=1 (2 bytes)
  g1[1] = (int)((tensor_d0 & 0xFFFFu) << 16);             // tensor_dim0[15:0]
  g1[2] = (int)(((tensor_d0 >> 16) & 0xFFFFu) |           // tensor_dim0[31:16]
                ((tensor_d1 & 0xFFFFu) << 16));           // tensor_dim1[15:0]
  g1[3] = (int)(((tensor_d1 >> 16) & 0xFFFFu) |           // tensor_dim1[31:16]
                ((tile_d0 & 0xFFFFu) << 16));             // tile_dim0
  g1[4] = (int)(tile_d1 & 0xFFFFu);                       // tile_dim1 (tile_dim2=0)
  g1[5] = (int)stride0;                                   // tensor_dim0_stride[31:0]
  g1[6] = 0;                                              // stride0[47:32], stride1[15:0]
  g1[7] = 0;
  i32x4 z4 = {0, 0, 0, 0};
#if defined(__clang_major__) && (__clang_major__ >= 23)
  i32x8 z8 = {};
  __builtin_amdgcn_tensor_load_to_lds(g0, g1, z4, z4, z8, 0);
#else
  __builtin_amdgcn_tensor_load_to_lds(g0, g1, z4, z4, 0);
#endif
}

// ---------------------------------------------------------------- kernel 0
// Convert fp32 weights to f16. wcat = [q_w; k_w; v_w] rows (1536 x 512).
__global__ void cvt_w_kernel(const float* __restrict__ qw, const float* __restrict__ kw,
                             const float* __restrict__ vw, const float* __restrict__ ow,
                             f16* __restrict__ wcat, f16* __restrict__ wout) {
  int i = blockIdx.x * 256 + threadIdx.x;     // grid 1024 * 256 = 512*512
  wcat[i]           = (f16)qw[i];
  wcat[F_*F_ + i]   = (f16)kw[i];
  wcat[2*F_*F_ + i] = (f16)vw[i];
  wout[i]           = (f16)ow[i];
}

// ---------------------------------------------------------------- kernel 1
// LayerNorm over F, output f16. One block (256 thr) per row.
__global__ void ln_kernel(const float* __restrict__ x, const float* __restrict__ gamma,
                          const float* __restrict__ beta, f16* __restrict__ xn) {
  int row = blockIdx.x;
  int tid = threadIdx.x;
  const float* xr = x + row * F_;
  float a0 = xr[tid], a1 = xr[tid + 256];
  float s = a0 + a1, ss = a0 * a0 + a1 * a1;
#pragma unroll
  for (int m = 1; m < 32; m <<= 1) {
    s  += __shfl_xor(s,  m, 32);
    ss += __shfl_xor(ss, m, 32);
  }
  __shared__ float smem[16];
  int wid = tid >> 5;
  if ((tid & 31) == 0) { smem[wid] = s; smem[8 + wid] = ss; }
  __syncthreads();
  float ts = 0.f, tss = 0.f;
#pragma unroll
  for (int w = 0; w < 8; ++w) { ts += smem[w]; tss += smem[8 + w]; }
  float mu  = ts * (1.0f / F_);
  float var = tss * (1.0f / F_) - mu * mu;
  float inv = rsqrtf(var + 1e-5f);
  f16* o = xn + row * F_;
  o[tid]       = (f16)((a0 - mu) * inv * gamma[tid]       + beta[tid]);
  o[tid + 256] = (f16)((a1 - mu) * inv * gamma[tid + 256] + beta[tid + 256]);
}

// ---------------------------------------------------------------- kernel 2
// QKV GEMM: xn(4096x512) x wcat(1536x512)^T. Each wave: 32(M) x 64(N) tile
// (2 A-frags, 8 accumulators, 8 WMMA per K-step).
// q -> (B,H,T,D), k -> (B,H,T,D), v -> (B,H,D,T). Bias order: (k, q, v).
__global__ void qkv_gemm_kernel(const f16* __restrict__ xn, const f16* __restrict__ wcat,
                                const float* __restrict__ bias,
                                f16* __restrict__ q4, f16* __restrict__ k4,
                                f16* __restrict__ v4t) {
  int wave  = (blockIdx.x * blockDim.x + threadIdx.x) >> 5;  // 0..3071
  int lane  = threadIdx.x & 31;
  int mtile = wave & 127;       // 128 tiles of 32 rows
  int ntile = wave >> 7;        // 0..23
  int sel   = lane >> 4;
  int l15   = lane & 15;
  int ncol0 = ntile * 64;

  v8f acc[2][4] = {};
  const f16* ar0 = xn + (mtile * 32 + l15) * F_;
  const f16* ar1 = ar0 + 16 * F_;

  for (int kk = 0; kk < 16; ++kk) {
    int kb = kk * 32;
    v16h a0 = load_frag(ar0 + kb + sel * 8, ar0 + kb + 16 + sel * 8);
    v16h a1 = load_frag(ar1 + kb + sel * 8, ar1 + kb + 16 + sel * 8);
    __builtin_prefetch(ar0 + kb + 64, 0, 3);   // global_prefetch_b8
    __builtin_prefetch(ar1 + kb + 64, 0, 3);
#pragma unroll
    for (int j = 0; j < 4; ++j) {
      int n = ncol0 + j * 16 + l15;
      const f16* brow = wcat + n * F_ + kb + sel * 8;
      v16h b = load_frag(brow, brow + 16);
      acc[0][j] = wmma_f16(a0, b, acc[0][j]);
      acc[1][j] = wmma_f16(a1, b, acc[1][j]);
    }
  }

#pragma unroll
  for (int j = 0; j < 4; ++j) {
    int n = ncol0 + j * 16 + l15;
    float bn;
    if (n < 512)       bn = bias[512 + n];    // q
    else if (n < 1024) bn = bias[n - 512];    // k
    else               bn = bias[n];          // v
    int nm = n & 511;
    int h = nm >> 6, d = nm & 63;
#pragma unroll
    for (int r = 0; r < 2; ++r)
#pragma unroll
      for (int e = 0; e < 8; ++e) {
        int m = mtile * 32 + r * 16 + e + sel * 8;
        int b = m >> 10, t = m & 1023;
        f16 val = (f16)(acc[r][j][e] + bn);
        if (n < 512)        q4[((b * H_ + h) * T_ + t) * D_ + d] = val;
        else if (n < 1024)  k4[((b * H_ + h) * T_ + t) * D_ + d] = val;
        else                v4t[((b * H_ + h) * D_ + d) * T_ + t] = val;
      }
  }
}

// ---------------------------------------------------------------- kernel 3
// small[bh*T + t, r] = q4[bh,t,:] . rel_emb[r,:]   (r = 0..64)
__global__ void rel_kernel(const f16* __restrict__ q4, const float* __restrict__ rel_emb,
                           float* __restrict__ small) {
  int row = blockIdx.x;          // 0 .. BH*T-1
  int r = threadIdx.x;
  if (r >= NREL) return;
  const f16*   qr = q4 + row * D_;
  const float* re = rel_emb + r * D_;
  float acc = 0.f;
#pragma unroll
  for (int d = 0; d < D_; ++d) acc += (float)qr[d] * re[d];
  small[row * NREL + r] = acc;
}

// ---------------------------------------------------------------- kernel 4
// Flash attention. 8 waves/block share one (b,h); K/V 32-key tiles staged
// into LDS once per block via TDM (double-buffered), 8x reuse across waves.
__global__ void attn_kernel(const f16* __restrict__ q4, const f16* __restrict__ k4,
                            const f16* __restrict__ v4t, const float* __restrict__ small,
                            const unsigned char* __restrict__ mask, f16* __restrict__ ctx) {
  __shared__ f16 kbuf[2][32 * 64];   // [s_local][d]
  __shared__ f16 vbuf[2][64 * 32];   // [d][s_local]
  __shared__ f16 pbuf[8][16 * 32];   // per-wave P bounce

  int waveId = threadIdx.x >> 5;
  int lane   = threadIdx.x & 31;
  int wave   = blockIdx.x * 8 + waveId;   // 0..4095
  int bh     = wave >> 6;                 // uniform across the block
  int rt     = wave & 63;
  int sel    = lane >> 4;
  int l15    = lane & 15;
  int b      = bh >> 3;
  int h      = bh & 7;

  const f16* qbase = q4 + (bh * T_ + rt * 16 + l15) * D_;
  v16h aq0 = load_frag(qbase + sel * 8,      qbase + 16 + sel * 8);
  v16h aq1 = load_frag(qbase + 32 + sel * 8, qbase + 48 + sel * 8);

  v8f o[4] = {};
  float mrow[8], lrow[8];
#pragma unroll
  for (int e = 0; e < 8; ++e) { mrow[e] = -3.0e38f; lrow[e] = 0.f; }

  f16* pl = pbuf[waveId];
  const f16* kslice = k4 + (size_t)bh * T_ * D_;   // (T x D) for this bh
  const f16* vslice = v4t + (size_t)bh * D_ * T_;  // (D x T) for this bh

  // Prologue: stage tile 0 via TDM (wave 0 only), then block barrier.
  if (waveId == 0) {
    tdm_load_2d((u32)(size_t)&kbuf[0][0], kslice, D_, T_, D_, D_, 32);
    tdm_load_2d((u32)(size_t)&vbuf[0][0], vslice, T_, D_, T_, 32, D_);
    __builtin_amdgcn_s_wait_tensorcnt((short)0);
  }
  __syncthreads();

  for (int st = 0; st < 32; ++st) {
    int s0 = st * 32;
    int cur = st & 1;
    // Prefetch next K/V tile into the other buffer.
    if (waveId == 0 && st + 1 < 32) {
      tdm_load_2d((u32)(size_t)&kbuf[cur ^ 1][0], kslice + (s0 + 32) * D_,
                  D_, T_, D_, D_, 32);
      tdm_load_2d((u32)(size_t)&vbuf[cur ^ 1][0], vslice + (s0 + 32),
                  T_, D_, T_, 32, D_);
    }

    const f16* kls = &kbuf[cur][0];
    const f16* vls = &vbuf[cur][0];

    v8f sc[2] = {};
#pragma unroll
    for (int sj = 0; sj < 2; ++sj) {
      const f16* kb = kls + (sj * 16 + l15) * D_;
      v16h b0 = load_frag(kb + sel * 8,      kb + 16 + sel * 8);
      sc[sj] = wmma_f16(aq0, b0, sc[sj]);
      v16h b1 = load_frag(kb + 32 + sel * 8, kb + 48 + sel * 8);
      sc[sj] = wmma_f16(aq1, b1, sc[sj]);
    }

    // rel-pos gather + key mask + 1/sqrt(D) scale; tile row max
    float tmax[8];
#pragma unroll
    for (int e = 0; e < 8; ++e) tmax[e] = -3.0e38f;
    float p[2][8];
#pragma unroll
    for (int sj = 0; sj < 2; ++sj) {
      int s = s0 + sj * 16 + l15;
      float madd = mask[b * T_ + s] ? 0.0f : -1.0e30f;
#pragma unroll
      for (int e = 0; e < 8; ++e) {
        int t = rt * 16 + e + sel * 8;
        int dist = s - t;
        dist = dist < -32 ? -32 : (dist > 32 ? 32 : dist);
        float v = (sc[sj][e] + small[(bh * T_ + t) * NREL + dist + 32] + madd) * 0.125f;
        p[sj][e] = v;
        tmax[e] = fmaxf(tmax[e], v);
      }
    }
    // row max across the 16 lanes holding each row (masks 1..8 stay in-half)
#pragma unroll
    for (int mk = 1; mk < 16; mk <<= 1)
#pragma unroll
      for (int e = 0; e < 8; ++e)
        tmax[e] = fmaxf(tmax[e], __shfl_xor(tmax[e], mk, 32));

    float rsum[8];
#pragma unroll
    for (int e = 0; e < 8; ++e) {
      float mnew  = fmaxf(mrow[e], tmax[e]);
      float alpha = __expf(mrow[e] - mnew);
      mrow[e] = mnew;
      lrow[e] *= alpha;
      rsum[e] = 0.f;
#pragma unroll
      for (int dt = 0; dt < 4; ++dt) o[dt][e] *= alpha;
    }
#pragma unroll
    for (int sj = 0; sj < 2; ++sj)
#pragma unroll
      for (int e = 0; e < 8; ++e) {
        float pe = __expf(p[sj][e] - mrow[e]);
        rsum[e] += pe;
        pl[(e + sel * 8) * 32 + sj * 16 + l15] = (f16)pe;  // P -> LDS (row, col)
      }
#pragma unroll
    for (int mk = 1; mk < 16; mk <<= 1)
#pragma unroll
      for (int e = 0; e < 8; ++e)
        rsum[e] += __shfl_xor(rsum[e], mk, 32);
#pragma unroll
    for (int e = 0; e < 8; ++e) lrow[e] += rsum[e];

    asm volatile("s_wait_dscnt 0" ::: "memory");     // wave-private LDS bounce

    // reload P as an A-fragment (16x32)
    const f16* pr = pl + l15 * 32;
    v16h pa = load_frag(pr + sel * 8, pr + 16 + sel * 8);

#pragma unroll
    for (int dt = 0; dt < 4; ++dt) {
      const f16* vb = vls + (dt * 16 + l15) * 32 + sel * 8;
      v16h bv = load_frag(vb, vb + 16);
      o[dt] = wmma_f16(pa, bv, o[dt]);
    }

    // Next tile must be resident before anyone proceeds.
    if (waveId == 0) __builtin_amdgcn_s_wait_tensorcnt((short)0);
    __syncthreads();
  }

  // normalize and write context (B*T, F) in f16
#pragma unroll
  for (int dt = 0; dt < 4; ++dt)
#pragma unroll
    for (int e = 0; e < 8; ++e) {
      float inv = lrow[e] > 0.f ? 1.0f / lrow[e] : 0.f;
      int t = rt * 16 + e + sel * 8;
      ctx[(b * T_ + t) * F_ + h * D_ + dt * 16 + l15] = (f16)(o[dt][e] * inv);
    }
}

// ---------------------------------------------------------------- kernel 5
// Output projection: ctx(4096x512) x out_w^T + out_b -> fp32. 32x64 per wave.
__global__ void out_gemm_kernel(const f16* __restrict__ ctx, const f16* __restrict__ wout,
                                const float* __restrict__ ob, float* __restrict__ out) {
  int wave  = (blockIdx.x * blockDim.x + threadIdx.x) >> 5;  // 0..1023
  int lane  = threadIdx.x & 31;
  int mtile = wave & 127;
  int ntile = wave >> 7;        // 0..7
  int sel   = lane >> 4;
  int l15   = lane & 15;
  int ncol0 = ntile * 64;

  v8f acc[2][4] = {};
  const f16* ar0 = ctx + (mtile * 32 + l15) * F_;
  const f16* ar1 = ar0 + 16 * F_;
  for (int kk = 0; kk < 16; ++kk) {
    int kb = kk * 32;
    v16h a0 = load_frag(ar0 + kb + sel * 8, ar0 + kb + 16 + sel * 8);
    v16h a1 = load_frag(ar1 + kb + sel * 8, ar1 + kb + 16 + sel * 8);
    __builtin_prefetch(ar0 + kb + 64, 0, 3);
#pragma unroll
    for (int j = 0; j < 4; ++j) {
      int n = ncol0 + j * 16 + l15;
      const f16* brow = wout + n * F_ + kb + sel * 8;
      v16h b = load_frag(brow, brow + 16);
      acc[0][j] = wmma_f16(a0, b, acc[0][j]);
      acc[1][j] = wmma_f16(a1, b, acc[1][j]);
    }
  }
#pragma unroll
  for (int j = 0; j < 4; ++j) {
    int n = ncol0 + j * 16 + l15;
    float bn = ob[n];
#pragma unroll
    for (int r = 0; r < 2; ++r)
#pragma unroll
      for (int e = 0; e < 8; ++e) {
        int m = mtile * 32 + r * 16 + e + sel * 8;
        out[m * F_ + n] = acc[r][j][e] + bn;
      }
  }
}

// ---------------------------------------------------------------- launcher
extern "C" void kernel_launch(void* const* d_in, const int* in_sizes, int n_in,
                              void* d_out, int out_size, void* d_ws, size_t ws_size,
                              hipStream_t stream) {
  const float* x      = (const float*)d_in[0];
  const unsigned char* seq_mask = (const unsigned char*)d_in[1];
  const float* gamma  = (const float*)d_in[2];
  const float* beta   = (const float*)d_in[3];
  const float* q_w    = (const float*)d_in[4];
  const float* k_w    = (const float*)d_in[5];
  const float* v_w    = (const float*)d_in[6];
  const float* bias   = (const float*)d_in[7];
  const float* relemb = (const float*)d_in[8];
  const float* out_w  = (const float*)d_in[9];
  const float* out_b  = (const float*)d_in[10];
  float* out = (float*)d_out;

  char* w = (char*)d_ws;
  f16* xn    = (f16*)w;  w += (size_t)BT * F_ * sizeof(f16);
  f16* wcat  = (f16*)w;  w += (size_t)3 * F_ * F_ * sizeof(f16);
  f16* wout  = (f16*)w;  w += (size_t)F_ * F_ * sizeof(f16);
  f16* q4    = (f16*)w;  w += (size_t)BH * T_ * D_ * sizeof(f16);
  f16* k4    = (f16*)w;  w += (size_t)BH * T_ * D_ * sizeof(f16);
  f16* v4t   = (f16*)w;  w += (size_t)BH * T_ * D_ * sizeof(f16);
  float* sml = (float*)w; w += (size_t)BH * T_ * NREL * sizeof(float);
  f16* ctx   = (f16*)w;  w += (size_t)BT * F_ * sizeof(f16);

  cvt_w_kernel<<<1024, 256, 0, stream>>>(q_w, k_w, v_w, out_w, wcat, wout);
  ln_kernel<<<BT, 256, 0, stream>>>(x, gamma, beta, xn);
  qkv_gemm_kernel<<<384, 256, 0, stream>>>(xn, wcat, bias, q4, k4, v4t);
  rel_kernel<<<BH * T_, 96, 0, stream>>>(q4, relemb, sml);
  attn_kernel<<<512, 256, 0, stream>>>(q4, k4, v4t, sml, seq_mask, ctx);
  out_gemm_kernel<<<128, 256, 0, stream>>>(ctx, wout, out_b, out);
}
